// SimpleTranslate_70437463654770
// MI455X (gfx1250) — compile-verified, matrix operand order checked by reading, not run on hardware
//
#include <hip/hip_runtime.h>
#include <hip/hip_bf16.h>
#include <math.h>

typedef __attribute__((ext_vector_type(16))) _Float16 v16h;
typedef __attribute__((ext_vector_type(8)))  float    v8f;
typedef _Float16 half_t;

// Model dims
#define Bq   4
#define Sq   512
#define Dm   512
#define Hh   8
#define DHh  64
#define Mf   2048
#define Ll   6
#define Vv   32000
#define NT   2048   // B*S tokens

// GEMM tiling
#define BM   64
#define BN   64
#define BKK  32
#define LDK  40     // padded LDS K-stride (f16): 80B rows -> 16B aligned chunks

// ---------------------------------------------------------------------------
// Weight packing: fp32 -> f16, stored as Bt[N][K] (row = output col, K inner)
// ---------------------------------------------------------------------------
__global__ void pack_qkv_kernel(const float* __restrict__ w /*[H,D,DH]*/,
                                half_t* __restrict__ wt /*[H*DH][D]*/) {
  int idx = blockIdx.x * blockDim.x + threadIdx.x;   // over 512*512
  if (idx >= Dm * Dm) return;
  int n = idx >> 9;        // n = h*64 + e
  int k = idx & 511;       // k = d
  int h = n >> 6, e = n & 63;
  wt[idx] = (half_t)w[((long)h * Dm + k) * DHh + e];
}

__global__ void pack_t_kernel(const float* __restrict__ w /*[K,N]*/,
                              half_t* __restrict__ wt /*[N][K]*/, int K, int N) {
  long idx = (long)blockIdx.x * blockDim.x + threadIdx.x;
  if (idx >= (long)K * N) return;
  int n = (int)(idx / K);
  int k = (int)(idx % K);
  wt[idx] = (half_t)w[(long)k * N + n];
}

// ---------------------------------------------------------------------------
// Embedding: x[row,:] = tok_emb[tok[row],:] + pos_emb[row%S,:]
// ---------------------------------------------------------------------------
__global__ void embed_kernel(const int* __restrict__ tok,
                             const float* __restrict__ te,
                             const float* __restrict__ pe,
                             float* __restrict__ x) {
  int row = blockIdx.x;              // b*S + s
  int s = row & (Sq - 1);
  int t = tok[row];
  for (int i = threadIdx.x; i < Dm; i += blockDim.x)
    x[(long)row * Dm + i] = te[(long)t * Dm + i] + pe[(long)s * Dm + i];
}

// ---------------------------------------------------------------------------
// LayerNorm (fp32 in, f16 out). One 256-thread block per row of 512.
// ---------------------------------------------------------------------------
__global__ void ln_kernel(const float* __restrict__ x,
                          const float* __restrict__ w,
                          const float* __restrict__ b,
                          half_t* __restrict__ out) {
  __shared__ float red[256];
  int row = blockIdx.x;
  int tid = threadIdx.x;
  const float* xr = x + (long)row * Dm;
  float v0 = xr[tid], v1 = xr[tid + 256];
  red[tid] = v0 + v1;
  __syncthreads();
  for (int st = 128; st > 0; st >>= 1) {
    if (tid < st) red[tid] += red[tid + st];
    __syncthreads();
  }
  float mu = red[0] * (1.0f / Dm);
  __syncthreads();
  float d0 = v0 - mu, d1 = v1 - mu;
  red[tid] = d0 * d0 + d1 * d1;
  __syncthreads();
  for (int st = 128; st > 0; st >>= 1) {
    if (tid < st) red[tid] += red[tid + st];
    __syncthreads();
  }
  float rstd = rsqrtf(red[0] * (1.0f / Dm) + 1e-5f);
  out[(long)row * Dm + tid]       = (half_t)(d0 * rstd * w[tid] + b[tid]);
  out[(long)row * Dm + tid + 256] = (half_t)(d1 * rstd * w[tid + 256] + b[tid + 256]);
}

// ---------------------------------------------------------------------------
// Masked softmax over scores row (len 512), scale fused. probs out as f16.
// grid: (s, h, b); block 256. causal=1 -> t<=s ; causal=0 -> srcTok!=0 mask.
// ---------------------------------------------------------------------------
__global__ void softmax_kernel(const float* __restrict__ scores,
                               half_t* __restrict__ probs,
                               const int* __restrict__ srcTok, int causal) {
  __shared__ float red[256];
  int s = blockIdx.x, h = blockIdx.y, b = blockIdx.z;
  int tid = threadIdx.x;
  long base = (((long)(b * Hh + h)) * Sq + s) * Sq;
  const float scale = 0.125f;   // 1/sqrt(64)
  float v[2];
  #pragma unroll
  for (int i = 0; i < 2; ++i) {
    int t = tid + i * 256;
    bool ok = causal ? (t <= s) : (srcTok[b * Sq + t] != 0);
    v[i] = ok ? scores[base + t] * scale : -1e30f;
  }
  red[tid] = fmaxf(v[0], v[1]);
  __syncthreads();
  for (int st = 128; st > 0; st >>= 1) {
    if (tid < st) red[tid] = fmaxf(red[tid], red[tid + st]);
    __syncthreads();
  }
  float mx = red[0];
  __syncthreads();
  float e0 = expf(v[0] - mx), e1 = expf(v[1] - mx);
  red[tid] = e0 + e1;
  __syncthreads();
  for (int st = 128; st > 0; st >>= 1) {
    if (tid < st) red[tid] += red[tid + st];
    __syncthreads();
  }
  float inv = 1.0f / red[0];
  probs[base + tid]       = (half_t)(e0 * inv);
  probs[base + tid + 256] = (half_t)(e1 * inv);
}

// ---------------------------------------------------------------------------
// Generic batched WMMA GEMM: C[z] = act(A[z] * Bt[z]^T + bias) + resid
//   A  : f16 [M,K] row-major, row stride lda, per-z offset zb*sAb + zh*sAh
//   Bt : f16 [N,K] row-major (K contiguous), stride ldb, per-z zb*sBb+zh*sBh
//   C  : fp32 (Cf) and/or f16 (Ch) row stride ldc, per-z zb*sCb+zh*sCh;
//        optional f16 transposed output ChT[col*ldct + row] (non-batched only)
//   z = blockIdx.z ; zb = z / nH ; zh = z % nH
// Block = 128 threads (4 wave32); 64x64 tile; each wave a 16-row strip.
// All global tile loads are 128-bit; LDS rows padded to 80B.
// ---------------------------------------------------------------------------
__global__ void __launch_bounds__(128)
gemm_wmma_kernel(const half_t* __restrict__ A, int lda, long sAb, long sAh,
                 const half_t* __restrict__ Bt, int ldb, long sBb, long sBh,
                 const float* __restrict__ bias,
                 const float* __restrict__ resid,
                 float* __restrict__ Cf, half_t* __restrict__ Ch,
                 half_t* __restrict__ ChT, int ldct,
                 int ldc, long sCb, long sCh,
                 int M, int N, int K, int nH, int act) {
  __shared__ __align__(16) half_t As[BM * LDK];
  __shared__ __align__(16) half_t Bs[BN * LDK];

  int tid  = threadIdx.x;
  int wave = tid >> 5;
  int lane = tid & 31;
  int m0 = blockIdx.y * BM;
  int n0 = blockIdx.x * BN;
  int z  = blockIdx.z;
  int zb = z / nH, zh = z % nH;
  const half_t* Ap = A  + zb * sAb + zh * sAh;
  const half_t* Bp = Bt + zb * sBb + zh * sBh;
  long coff = zb * sCb + zh * sCh;

  v8f acc[4];
  #pragma unroll
  for (int nb = 0; nb < 4; ++nb)
    #pragma unroll
    for (int r = 0; r < 8; ++r) acc[nb][r] = 0.0f;

  int mr = lane & 15;     // M (or N) index within 16
  int hf = lane >> 4;     // lane half selects K group

  for (int k0 = 0; k0 < K; k0 += BKK) {
    // ---- stage 64x32 A and Bt tiles into LDS with 128-bit loads ----
    // 64x32 f16 = 256 chunks of 8 f16; 128 threads -> 2 chunks each per tile
    #pragma unroll
    for (int i = 0; i < 2; ++i) {
      int idx = tid + i * 128;           // 0..255
      int r  = idx >> 2;
      int c8 = (idx & 3) * 8;
      *(uint4*)&As[r * LDK + c8] =
          *(const uint4*)&Ap[(long)(m0 + r) * lda + (k0 + c8)];
      *(uint4*)&Bs[r * LDK + c8] =
          *(const uint4*)&Bp[(long)(n0 + r) * ldb + (k0 + c8)];
    }
    if (k0 + BKK < K)
      __builtin_prefetch(Ap + (long)m0 * lda + (k0 + BKK), 0, 1);
    __syncthreads();

    // ---- A fragment: 16x32, ISA 16-bit A layout ----
    union { v16h v; unsigned u[8]; } af;
    const unsigned* As32 = (const unsigned*)&As[(wave * 16 + mr) * LDK];
    #pragma unroll
    for (int vv = 0; vv < 4; ++vv) {
      af.u[vv]     = As32[hf * 4 + vv];        // K = hf*8 + 2vv, +1
      af.u[vv + 4] = As32[8 + hf * 4 + vv];    // K = 16 + hf*8 + 2vv, +1
    }

    // ---- 4 WMMAs: one per 16-col block ----
    #pragma unroll
    for (int nb = 0; nb < 4; ++nb) {
      union { v16h v; unsigned u[8]; } bf;
      const unsigned* Bs32 = (const unsigned*)&Bs[(nb * 16 + mr) * LDK];
      #pragma unroll
      for (int vv = 0; vv < 4; ++vv) {
        bf.u[vv]     = Bs32[hf * 4 + vv];
        bf.u[vv + 4] = Bs32[8 + hf * 4 + vv];
      }
      acc[nb] = __builtin_amdgcn_wmma_f32_16x16x32_f16(
          false, af.v, false, bf.v, (short)0, acc[nb], false, false);
    }
    __syncthreads();
  }

  // ---- epilogue: bias, gelu, residual; C layout: VGPR r -> M = r + 8*hf ----
  #pragma unroll
  for (int nb = 0; nb < 4; ++nb) {
    int col = n0 + nb * 16 + mr;
    float bv = bias ? bias[col] : 0.0f;
    #pragma unroll
    for (int r = 0; r < 8; ++r) {
      int row = m0 + wave * 16 + hf * 8 + r;
      float v = acc[nb][r] + bv;
      if (act) v = 0.5f * v * (1.0f + erff(v * 0.70710678118654752f));
      if (resid) v += resid[(long)row * ldc + col];
      long ci = coff + (long)row * ldc + col;
      if (Cf) Cf[ci] = v;
      if (Ch) Ch[ci] = (half_t)v;
      if (ChT) ChT[(long)col * ldct + row] = (half_t)v;
    }
  }
}

// ---------------------------------------------------------------------------
// Host orchestration
// ---------------------------------------------------------------------------
extern "C" void kernel_launch(void* const* d_in, const int* in_sizes, int n_in,
                              void* d_out, int out_size, void* d_ws, size_t ws_size,
                              hipStream_t stream) {
  (void)in_sizes; (void)n_in; (void)out_size; (void)ws_size;
  const int*   tokS      = (const int*)d_in[0];
  const int*   tokD      = (const int*)d_in[1];
  const float* tok_emb   = (const float*)d_in[2];
  const float* pos_emb   = (const float*)d_in[3];
  const float* enc_ln1_w = (const float*)d_in[4];
  const float* enc_ln1_b = (const float*)d_in[5];
  const float* enc_wq    = (const float*)d_in[6];
  const float* enc_wk    = (const float*)d_in[7];
  const float* enc_wv    = (const float*)d_in[8];
  const float* enc_wo    = (const float*)d_in[9];
  const float* enc_bo    = (const float*)d_in[10];
  const float* enc_ln2_w = (const float*)d_in[11];
  const float* enc_ln2_b = (const float*)d_in[12];
  const float* enc_w1    = (const float*)d_in[13];
  const float* enc_b1    = (const float*)d_in[14];
  const float* enc_w2    = (const float*)d_in[15];
  const float* enc_b2    = (const float*)d_in[16];
  const float* enc_lnf_w = (const float*)d_in[17];
  const float* enc_lnf_b = (const float*)d_in[18];
  const float* dec_ln1_w = (const float*)d_in[19];
  const float* dec_ln1_b = (const float*)d_in[20];
  const float* dec_wq_s  = (const float*)d_in[21];
  const float* dec_wk_s  = (const float*)d_in[22];
  const float* dec_wv_s  = (const float*)d_in[23];
  const float* dec_wo_s  = (const float*)d_in[24];
  const float* dec_bo_s  = (const float*)d_in[25];
  const float* dec_ln2_w = (const float*)d_in[26];
  const float* dec_ln2_b = (const float*)d_in[27];
  const float* dec_wq_c  = (const float*)d_in[28];
  const float* dec_wk_c  = (const float*)d_in[29];
  const float* dec_wv_c  = (const float*)d_in[30];
  const float* dec_wo_c  = (const float*)d_in[31];
  const float* dec_bo_c  = (const float*)d_in[32];
  const float* dec_ln3_w = (const float*)d_in[33];
  const float* dec_ln3_b = (const float*)d_in[34];
  const float* dec_w1    = (const float*)d_in[35];
  const float* dec_b1    = (const float*)d_in[36];
  const float* dec_w2    = (const float*)d_in[37];
  const float* dec_b2    = (const float*)d_in[38];
  const float* dec_lnf_w = (const float*)d_in[39];
  const float* dec_lnf_b = (const float*)d_in[40];
  const float* head_w    = (const float*)d_in[41];
  const float* head_b    = (const float*)d_in[42];
  float* out = (float*)d_out;

  char* ws = (char*)d_ws;
  size_t off = 0;
  auto alloc = [&](size_t bytes) -> char* {
    size_t a = (off + 255) & ~(size_t)255;
    off = a + bytes;
    return ws + a;
  };

  const dim3 blk256(256);
  const int gqkv = (Dm * Dm + 255) / 256;

  // ---- pack all weights to f16 Bt layout ----
  half_t *eWq[Ll], *eWk[Ll], *eWv[Ll], *eWo[Ll], *eW1[Ll], *eW2[Ll];
  half_t *dWqS[Ll], *dWkS[Ll], *dWvS[Ll], *dWoS[Ll];
  half_t *dWqC[Ll], *dWkC[Ll], *dWvC[Ll], *dWoC[Ll];
  half_t *dW1[Ll], *dW2[Ll];
  for (int i = 0; i < Ll; ++i) {
    long o44 = (long)i * Dm * Dm;       // H*D*DH == 512*512
    long oDM = (long)i * Dm * Mf;
    eWq[i] = (half_t*)alloc((size_t)Dm * Dm * 2);
    pack_qkv_kernel<<<gqkv, blk256, 0, stream>>>(enc_wq + o44, eWq[i]);
    eWk[i] = (half_t*)alloc((size_t)Dm * Dm * 2);
    pack_qkv_kernel<<<gqkv, blk256, 0, stream>>>(enc_wk + o44, eWk[i]);
    eWv[i] = (half_t*)alloc((size_t)Dm * Dm * 2);
    pack_qkv_kernel<<<gqkv, blk256, 0, stream>>>(enc_wv + o44, eWv[i]);
    eWo[i] = (half_t*)alloc((size_t)Dm * Dm * 2);
    pack_t_kernel<<<gqkv, blk256, 0, stream>>>(enc_wo + o44, eWo[i], Dm, Dm);
    eW1[i] = (half_t*)alloc((size_t)Dm * Mf * 2);
    pack_t_kernel<<<(Dm * Mf + 255) / 256, blk256, 0, stream>>>(enc_w1 + oDM, eW1[i], Dm, Mf);
    eW2[i] = (half_t*)alloc((size_t)Mf * Dm * 2);
    pack_t_kernel<<<(Mf * Dm + 255) / 256, blk256, 0, stream>>>(enc_w2 + oDM, eW2[i], Mf, Dm);

    dWqS[i] = (half_t*)alloc((size_t)Dm * Dm * 2);
    pack_qkv_kernel<<<gqkv, blk256, 0, stream>>>(dec_wq_s + o44, dWqS[i]);
    dWkS[i] = (half_t*)alloc((size_t)Dm * Dm * 2);
    pack_qkv_kernel<<<gqkv, blk256, 0, stream>>>(dec_wk_s + o44, dWkS[i]);
    dWvS[i] = (half_t*)alloc((size_t)Dm * Dm * 2);
    pack_qkv_kernel<<<gqkv, blk256, 0, stream>>>(dec_wv_s + o44, dWvS[i]);
    dWoS[i] = (half_t*)alloc((size_t)Dm * Dm * 2);
    pack_t_kernel<<<gqkv, blk256, 0, stream>>>(dec_wo_s + o44, dWoS[i], Dm, Dm);
    dWqC[i] = (half_t*)alloc((size_t)Dm * Dm * 2);
    pack_qkv_kernel<<<gqkv, blk256, 0, stream>>>(dec_wq_c + o44, dWqC[i]);
    dWkC[i] = (half_t*)alloc((size_t)Dm * Dm * 2);
    pack_qkv_kernel<<<gqkv, blk256, 0, stream>>>(dec_wk_c + o44, dWkC[i]);
    dWvC[i] = (half_t*)alloc((size_t)Dm * Dm * 2);
    pack_qkv_kernel<<<gqkv, blk256, 0, stream>>>(dec_wv_c + o44, dWvC[i]);
    dWoC[i] = (half_t*)alloc((size_t)Dm * Dm * 2);
    pack_t_kernel<<<gqkv, blk256, 0, stream>>>(dec_wo_c + o44, dWoC[i], Dm, Dm);
    dW1[i] = (half_t*)alloc((size_t)Dm * Mf * 2);
    pack_t_kernel<<<(Dm * Mf + 255) / 256, blk256, 0, stream>>>(dec_w1 + oDM, dW1[i], Dm, Mf);
    dW2[i] = (half_t*)alloc((size_t)Mf * Dm * 2);
    pack_t_kernel<<<(Mf * Dm + 255) / 256, blk256, 0, stream>>>(dec_w2 + oDM, dW2[i], Mf, Dm);
  }
  half_t* Whead = (half_t*)alloc((size_t)Vv * Dm * 2);
  pack_t_kernel<<<((long)Vv * Dm + 255) / 256, blk256, 0, stream>>>(head_w, Whead, Dm, Vv);

  // ---- activations ----
  float*  x    = (float*)alloc((size_t)NT * Dm * 4);
  float*  y    = (float*)alloc((size_t)NT * Dm * 4);
  half_t* hbuf = (half_t*)alloc((size_t)NT * Dm * 2);
  half_t* qb   = (half_t*)alloc((size_t)NT * Dm * 2);
  half_t* kb   = (half_t*)alloc((size_t)NT * Dm * 2);
  half_t* vT   = (half_t*)alloc((size_t)Dm * NT * 2);  // v transposed [D, B*S]
  half_t* ob   = (half_t*)alloc((size_t)NT * Dm * 2);
  half_t* xe   = (half_t*)alloc((size_t)NT * Dm * 2);
  half_t* mid  = (half_t*)alloc((size_t)NT * Mf * 2);
  float*  sc   = (float*)alloc((size_t)Bq * Hh * Sq * Sq * 4);
  half_t* pr   = (half_t*)alloc((size_t)Bq * Hh * Sq * Sq * 2);

  auto gemm = [&](const half_t* A, int lda, long sAb, long sAh,
                  const half_t* Bm, int ldb, long sBb, long sBh,
                  const float* bias, const float* resid,
                  float* Cf, half_t* Ch, half_t* ChT, int ldct,
                  int ldc, long sCb, long sCh,
                  int M, int N, int K, int nZ, int nH, int act) {
    dim3 grid(N / BN, M / BM, nZ);
    gemm_wmma_kernel<<<grid, dim3(128), 0, stream>>>(
        A, lda, sAb, sAh, Bm, ldb, sBb, sBh, bias, resid,
        Cf, Ch, ChT, ldct, ldc, sCb, sCh, M, N, K, nH, act);
  };

  const long sBH = (long)Sq * Dm;          // 262144: (b) stride in q/k/o
  const long sSC = (long)Sq * Sq;          // 262144: per-(b,h) score block
  const long sVTb = Sq;                    // vT (b) stride: 512
  const long sVTh = (long)DHh * NT;        // vT (h) stride: 131072
  const dim3 smGrid(Sq, Hh, Bq);

  // =================== Encoder ===================
  embed_kernel<<<NT, blk256, 0, stream>>>(tokS, tok_emb, pos_emb, x);
  for (int i = 0; i < Ll; ++i) {
    ln_kernel<<<NT, blk256, 0, stream>>>(x, enc_ln1_w + i * Dm, enc_ln1_b + i * Dm, hbuf);
    gemm(hbuf, Dm, 0, 0, eWq[i], Dm, 0, 0, nullptr, nullptr,
         nullptr, qb, nullptr, 0, Dm, 0, 0, NT, Dm, Dm, 1, 1, 0);
    gemm(hbuf, Dm, 0, 0, eWk[i], Dm, 0, 0, nullptr, nullptr,
         nullptr, kb, nullptr, 0, Dm, 0, 0, NT, Dm, Dm, 1, 1, 0);
    gemm(hbuf, Dm, 0, 0, eWv[i], Dm, 0, 0, nullptr, nullptr,
         nullptr, nullptr, vT, NT, Dm, 0, 0, NT, Dm, Dm, 1, 1, 0);
    // scores[b,h] = q[b,:,h,:] @ k[b,:,h,:]^T   (M=N=512, K=64)
    gemm(qb, Dm, sBH, DHh, kb, Dm, sBH, DHh, nullptr, nullptr,
         sc, nullptr, nullptr, 0, Sq, Hh * sSC, sSC, Sq, Sq, DHh, Bq * Hh, Hh, 0);
    softmax_kernel<<<smGrid, blk256, 0, stream>>>(sc, pr, tokS, 0);
    // o[b,:,h,:] = probs[b,h] @ v[b,:,h,:]     (M=512, N=64, K=512), B = vT
    gemm(pr, Sq, Hh * sSC, sSC, vT, NT, sVTb, sVTh, nullptr, nullptr,
         nullptr, ob, nullptr, 0, Dm, sBH, DHh, Sq, DHh, Sq, Bq * Hh, Hh, 0);
    gemm(ob, Dm, 0, 0, eWo[i], Dm, 0, 0, enc_bo + i * Dm, x,
         x, nullptr, nullptr, 0, Dm, 0, 0, NT, Dm, Dm, 1, 1, 0);
    ln_kernel<<<NT, blk256, 0, stream>>>(x, enc_ln2_w + i * Dm, enc_ln2_b + i * Dm, hbuf);
    gemm(hbuf, Dm, 0, 0, eW1[i], Dm, 0, 0, enc_b1 + i * Mf, nullptr,
         nullptr, mid, nullptr, 0, Mf, 0, 0, NT, Mf, Dm, 1, 1, 1);
    gemm(mid, Mf, 0, 0, eW2[i], Mf, 0, 0, enc_b2 + i * Dm, x,
         x, nullptr, nullptr, 0, Dm, 0, 0, NT, Dm, Mf, 1, 1, 0);
  }
  ln_kernel<<<NT, blk256, 0, stream>>>(x, enc_lnf_w, enc_lnf_b, xe);

  // =================== Decoder ===================
  embed_kernel<<<NT, blk256, 0, stream>>>(tokD, tok_emb, pos_emb, y);
  for (int i = 0; i < Ll; ++i) {
    // --- causal self-attention ---
    ln_kernel<<<NT, blk256, 0, stream>>>(y, dec_ln1_w + i * Dm, dec_ln1_b + i * Dm, hbuf);
    gemm(hbuf, Dm, 0, 0, dWqS[i], Dm, 0, 0, nullptr, nullptr,
         nullptr, qb, nullptr, 0, Dm, 0, 0, NT, Dm, Dm, 1, 1, 0);
    gemm(hbuf, Dm, 0, 0, dWkS[i], Dm, 0, 0, nullptr, nullptr,
         nullptr, kb, nullptr, 0, Dm, 0, 0, NT, Dm, Dm, 1, 1, 0);
    gemm(hbuf, Dm, 0, 0, dWvS[i], Dm, 0, 0, nullptr, nullptr,
         nullptr, nullptr, vT, NT, Dm, 0, 0, NT, Dm, Dm, 1, 1, 0);
    gemm(qb, Dm, sBH, DHh, kb, Dm, sBH, DHh, nullptr, nullptr,
         sc, nullptr, nullptr, 0, Sq, Hh * sSC, sSC, Sq, Sq, DHh, Bq * Hh, Hh, 0);
    softmax_kernel<<<smGrid, blk256, 0, stream>>>(sc, pr, tokS, 1);
    gemm(pr, Sq, Hh * sSC, sSC, vT, NT, sVTb, sVTh, nullptr, nullptr,
         nullptr, ob, nullptr, 0, Dm, sBH, DHh, Sq, DHh, Sq, Bq * Hh, Hh, 0);
    gemm(ob, Dm, 0, 0, dWoS[i], Dm, 0, 0, dec_bo_s + i * Dm, y,
         y, nullptr, nullptr, 0, Dm, 0, 0, NT, Dm, Dm, 1, 1, 0);
    // --- cross-attention (KV from encoder output xe) ---
    ln_kernel<<<NT, blk256, 0, stream>>>(y, dec_ln2_w + i * Dm, dec_ln2_b + i * Dm, hbuf);
    gemm(hbuf, Dm, 0, 0, dWqC[i], Dm, 0, 0, nullptr, nullptr,
         nullptr, qb, nullptr, 0, Dm, 0, 0, NT, Dm, Dm, 1, 1, 0);
    gemm(xe,   Dm, 0, 0, dWkC[i], Dm, 0, 0, nullptr, nullptr,
         nullptr, kb, nullptr, 0, Dm, 0, 0, NT, Dm, Dm, 1, 1, 0);
    gemm(xe,   Dm, 0, 0, dWvC[i], Dm, 0, 0, nullptr, nullptr,
         nullptr, nullptr, vT, NT, Dm, 0, 0, NT, Dm, Dm, 1, 1, 0);
    gemm(qb, Dm, sBH, DHh, kb, Dm, sBH, DHh, nullptr, nullptr,
         sc, nullptr, nullptr, 0, Sq, Hh * sSC, sSC, Sq, Sq, DHh, Bq * Hh, Hh, 0);
    softmax_kernel<<<smGrid, blk256, 0, stream>>>(sc, pr, tokS, 0);
    gemm(pr, Sq, Hh * sSC, sSC, vT, NT, sVTb, sVTh, nullptr, nullptr,
         nullptr, ob, nullptr, 0, Dm, sBH, DHh, Sq, DHh, Sq, Bq * Hh, Hh, 0);
    gemm(ob, Dm, 0, 0, dWoC[i], Dm, 0, 0, dec_bo_c + i * Dm, y,
         y, nullptr, nullptr, 0, Dm, 0, 0, NT, Dm, Dm, 1, 1, 0);
    // --- FFN ---
    ln_kernel<<<NT, blk256, 0, stream>>>(y, dec_ln3_w + i * Dm, dec_ln3_b + i * Dm, hbuf);
    gemm(hbuf, Dm, 0, 0, dW1[i], Dm, 0, 0, dec_b1 + i * Mf, nullptr,
         nullptr, mid, nullptr, 0, Mf, 0, 0, NT, Mf, Dm, 1, 1, 1);
    gemm(mid, Mf, 0, 0, dW2[i], Mf, 0, 0, dec_b2 + i * Dm, y,
         y, nullptr, nullptr, 0, Dm, 0, 0, NT, Dm, Mf, 1, 1, 0);
  }
  ln_kernel<<<NT, blk256, 0, stream>>>(y, dec_lnf_w, dec_lnf_b, hbuf);

  // =================== LM head ===================
  gemm(hbuf, Dm, 0, 0, Whead, Dm, 0, 0, head_b, nullptr,
       out, nullptr, nullptr, 0, Vv, 0, 0, NT, Vv, Dm, 1, 1, 0);
}
